// MultiHeadAttention_21139829031674
// MI455X (gfx1250) — matrix operand order repaired and checked by reading, not compiled, not run on hardware
//
#include <hip/hip_runtime.h>
#include <hip/hip_bf16.h>

#define HEADS 12
#define DM    768
#define DK    64
#define SEQ   2048
#define BATCH 2
#define MROWS (BATCH * SEQ)   // 4096

typedef __attribute__((ext_vector_type(16))) __bf16 v16bf;
typedef __attribute__((ext_vector_type(8)))  __bf16 v8bf;
typedef __attribute__((ext_vector_type(2)))  __bf16 v2bf;
typedef __attribute__((ext_vector_type(8)))  float  v8f;
typedef __attribute__((ext_vector_type(4)))  int    v4i;

#define AS1 __attribute__((address_space(1)))
#define AS3 __attribute__((address_space(3)))

#if defined(__gfx1250__) && __has_builtin(__builtin_amdgcn_global_load_async_to_lds_b128)
#define HAS_ASYNC_LDS 1
#else
#define HAS_ASYNC_LDS 0
#endif

// 16-byte global->LDS copy; async (ASYNCcnt-tracked) when available.
static __device__ inline void copy16_g2l(const __bf16* g, __bf16* l) {
#if HAS_ASYNC_LDS
  __builtin_amdgcn_global_load_async_to_lds_b128(
      (AS1 v4i*)g, (AS3 v4i*)l, 0, 0);
#else
  *(v8bf*)l = *(const v8bf*)g;
#endif
}
static __device__ inline void async_join() {
#if HAS_ASYNC_LDS
  asm volatile("s_wait_asynccnt 0" ::: "memory");
#endif
}

static __device__ inline v8f wmma_bf16(v16bf a, v16bf b, v8f c) {
  return __builtin_amdgcn_wmma_f32_16x16x32_bf16(
      false, a, false, b, (short)0, c, false, false);
}

// ---- Fragment loaders (CDNA5 wave32 WMMA layouts) -------------------------
// A 16x32 (MxK) bf16: lane L -> row M = L&15, half = L>>4;
//   elems 0..15 = base[row*ld + half*16 .. +15].
static __device__ inline v16bf load_a_tile(const __bf16* base, int ld) {
  const int lane = threadIdx.x & 31, row = lane & 15, half = lane >> 4;
  const __bf16* p = base + row * ld + half * 16;
  const v8bf lo = *(const v8bf*)(p);
  const v8bf hi = *(const v8bf*)(p + 8);
  return __builtin_shufflevector(lo, hi, 0, 1, 2, 3, 4, 5, 6, 7,
                                 8, 9, 10, 11, 12, 13, 14, 15);
}
// B 32x16 (KxN) bf16 with B(k,n) = base[n*ld + k]: lane L -> col N = L&15,
// 16 consecutive K at (L>>4)*16 -> single 32-byte load.
static __device__ inline v16bf load_b_tile(const __bf16* base, int ld) {
  const int lane = threadIdx.x & 31, n = lane & 15, kb = (lane >> 4) << 4;
  return *(const v16bf*)(base + n * ld + kb);
}
// C/D 16x16 f32: lane L -> col N = L&15; vgpr r -> row M = r + 8*(L>>4)

// ---- Register-staged f32 tile (ROWS x 32), 128 threads, fully unrolled ----
template <int ROWS>
struct TileRegs { float2 d[ROWS * 32 / 256]; };

template <int ROWS>
static __device__ inline void tile_gload(const float* __restrict__ src,
                                         size_t ld, TileRegs<ROWS>& t) {
  const int tid = threadIdx.x;
#pragma unroll
  for (int j = 0; j < ROWS * 32 / 256; ++j) {     // all loads issue together
    const int i = tid * 2 + j * 256;
    const int r = i >> 5, c = i & 31;
    t.d[j] = *(const float2*)(src + (size_t)r * ld + c);
  }
}
template <int ROWS>
static __device__ inline void tile_lstore(__bf16* dst, const TileRegs<ROWS>& t) {
  const int tid = threadIdx.x;
#pragma unroll
  for (int j = 0; j < ROWS * 32 / 256; ++j) {
    const int i = tid * 2 + j * 256;
    v2bf p;
    p[0] = (__bf16)t.d[j].x;                      // v_cvt_pk_bf16_f32
    p[1] = (__bf16)t.d[j].y;
    *(v2bf*)(dst + i) = p;
  }
}

// ============================================================================
// Shared GEMM body: C(128x64) = A(128xK) * W(64xK)^T, bf16 WMMA, f32 accum.
// Software-pipelined: next K-tile global loads issue before this tile's WMMAs.
// ============================================================================
template <typename StoreFn>
static __device__ inline void gemm_body(const float* __restrict__ A,
                                        const float* __restrict__ W,
                                        int m0, int n0, __bf16* lA, __bf16* lW,
                                        StoreFn&& store) {
  const int tid = threadIdx.x, wave = tid >> 5;
  TileRegs<128> ra;
  TileRegs<64> rw;
  tile_gload(A + (size_t)m0 * DM, DM, ra);
  tile_gload(W + (size_t)n0 * DM, DM, rw);

  v8f acc[2][4] = {};
  for (int k0 = 0; k0 < DM; k0 += 32) {
    tile_lstore(lA, ra);
    tile_lstore(lW, rw);
    __syncthreads();
    if (k0 + 32 < DM) {          // issue next tile's loads; wait is deferred
      tile_gload(A + (size_t)m0 * DM + k0 + 32, DM, ra);
      tile_gload(W + (size_t)n0 * DM + k0 + 32, DM, rw);
    }
    const v16bf a0 = load_a_tile(lA + (wave * 32) * 32, 32);
    const v16bf a1 = load_a_tile(lA + (wave * 32 + 16) * 32, 32);
#pragma unroll
    for (int nt = 0; nt < 4; ++nt) {
      const v16bf b = load_b_tile(lW + nt * 16 * 32, 32);
      acc[0][nt] = wmma_bf16(a0, b, acc[0][nt]);
      acc[1][nt] = wmma_bf16(a1, b, acc[1][nt]);
    }
    __syncthreads();
  }
  store(acc);
}

// ============================================================================
// Kernel 1: fused q/k/v projection  Y = X @ Wk^T + bk, bf16 out [B,H,S,Dk]
// grid (M/128, N/64, 3), block 128 (4 waves).
// ============================================================================
__global__ __launch_bounds__(128)
void proj_qkv_kernel(const float* __restrict__ q, const float* __restrict__ k,
                     const float* __restrict__ v, const float* __restrict__ W,
                     const float* __restrict__ bias,
                     __bf16* __restrict__ oq, __bf16* __restrict__ ok,
                     __bf16* __restrict__ ov) {
  __shared__ __bf16 lA[128 * 32];
  __shared__ __bf16 lW[64 * 32];
  const int tid = threadIdx.x, wave = tid >> 5;
  const int lane = tid & 31, nn = lane & 15, half = lane >> 4;
  const int m0 = blockIdx.x * 128;
  const int n0 = blockIdx.y * 64;
  const float* X = (blockIdx.z == 0) ? q : (blockIdx.z == 1) ? k : v;
  __bf16* O = (blockIdx.z == 0) ? oq : (blockIdx.z == 1) ? ok : ov;

  gemm_body(X, W, m0, n0, lA, lW, [&](v8f acc[2][4]) {
#pragma unroll
    for (int mt = 0; mt < 2; ++mt)
#pragma unroll
      for (int nt = 0; nt < 4; ++nt)
#pragma unroll
        for (int r = 0; r < 8; ++r) {
          const int gm = m0 + wave * 32 + mt * 16 + r + half * 8;
          const int gn = n0 + nt * 16 + nn;
          const int bb = gm >> 11, s = gm & (SEQ - 1);
          const int h = gn >> 6, d = gn & (DK - 1);
          const float val = acc[mt][nt][r] + bias[gn];
          O[(((size_t)bb * HEADS + h) * SEQ + s) * DK + d] = (__bf16)val;
        }
  });
}

// ============================================================================
// Kernel 2: flash attention per (b,h).  grid (SEQ/64, B*H), block 128.
// Double-buffered 64-key tiles: async K-DMA + V reg-loads for tile n+1 issue
// before tile n's WMMAs; one barrier per tile.
// ============================================================================
__global__ __launch_bounds__(128)
void attn_kernel(const __bf16* __restrict__ qh, const __bf16* __restrict__ kh,
                 const __bf16* __restrict__ vh, float* __restrict__ concat) {
  __shared__ __bf16 lK[2][64 * 64];    // [key][dim], 8 KB each
  __shared__ __bf16 lVt[2][64 * 64];   // [dim][key], 8 KB each
  __shared__ __bf16 lP[4][16 * 64];    // per-wave P [query][key], 8 KB
  const int tid = threadIdx.x, wave = tid >> 5, lane = tid & 31;
  const int half = lane >> 4, nn = lane & 15;
  const int bh = blockIdx.y;
  const int b = bh / HEADS, h = bh % HEADS;
  const int q0 = blockIdx.x * 64;
  const __bf16* Q = qh + (size_t)bh * SEQ * DK;
  const __bf16* K = kh + (size_t)bh * SEQ * DK;
  const __bf16* V = vh + (size_t)bh * SEQ * DK;

  const v16bf qa0 = load_a_tile(Q + (size_t)(q0 + wave * 16) * DK, DK);
  const v16bf qa1 = load_a_tile(Q + (size_t)(q0 + wave * 16) * DK + 32, DK);

  v8bf rv[4];  // staged V chunks (this thread's 4 x 8 dims)
  auto asyncK = [&](int t0, __bf16* dst) {
#pragma unroll
    for (int j = 0; j < 4; ++j) {
      const int i = (tid + j * 128) * 8;
      copy16_g2l(K + (size_t)t0 * DK + i, dst + i);
    }
  };
  auto loadV = [&](int t0) {
#pragma unroll
    for (int j = 0; j < 4; ++j) {
      const int c = tid + j * 128;
      rv[j] = *(const v8bf*)(V + (size_t)(t0 + (c >> 3)) * DK + (c & 7) * 8);
    }
  };
  auto scatterV = [&](__bf16* dst) {
#pragma unroll
    for (int j = 0; j < 4; ++j) {
      const int t = (tid + j * 128) >> 3, d0 = ((tid + j * 128) & 7) * 8;
#pragma unroll
      for (int e = 0; e < 8; ++e) dst[(d0 + e) * 64 + t] = rv[j][e];
    }
  };

  float mrow[8], lrow[8];
  v8f o[4] = {};
#pragma unroll
  for (int r = 0; r < 8; ++r) { mrow[r] = -1e30f; lrow[r] = 0.0f; }

  // prologue: stage tile 0
  asyncK(0, lK[0]);
  loadV(0);
  scatterV(lVt[0]);
  async_join();
  __syncthreads();

  int buf = 0;
  for (int t0 = 0; t0 < SEQ; t0 += 64, buf ^= 1) {
    const bool more = (t0 + 64) < SEQ;
    if (more) {                       // stage tile n+1 under tile n's compute
      asyncK(t0 + 64, lK[buf ^ 1]);
      loadV(t0 + 64);
    }

    // scores: S[16q x 64k] as four 16x16 C-fragments
    v8f s[4] = {};
#pragma unroll
    for (int g = 0; g < 4; ++g) {
      s[g] = wmma_bf16(qa0, load_b_tile(lK[buf] + g * 16 * 64, 64), s[g]);
      s[g] = wmma_bf16(qa1, load_b_tile(lK[buf] + g * 16 * 64 + 32, 64), s[g]);
    }

    // online softmax: row = r + 8*half lives in the 16 lanes of this half
#pragma unroll
    for (int r = 0; r < 8; ++r) {
      float x[4];
      float mx = -1e30f;
#pragma unroll
      for (int g = 0; g < 4; ++g) {
        x[g] = s[g][r] * 0.125f;       // 1/sqrt(64)
        mx = fmaxf(mx, x[g]);
      }
      mx = fmaxf(mx, __shfl_xor(mx, 1, 32));
      mx = fmaxf(mx, __shfl_xor(mx, 2, 32));
      mx = fmaxf(mx, __shfl_xor(mx, 4, 32));
      mx = fmaxf(mx, __shfl_xor(mx, 8, 32));
      const float mnew = fmaxf(mrow[r], mx);
      const float alpha = __expf(mrow[r] - mnew);
      float ps = 0.0f;
      const int prow = (r + half * 8) * 64 + nn;
#pragma unroll
      for (int g = 0; g < 4; ++g) {
        const float p = __expf(x[g] - mnew);
        ps += p;
        lP[wave][prow + 16 * g] = (__bf16)p;  // wave-private, same-wave DS
      }
      ps += __shfl_xor(ps, 1, 32);
      ps += __shfl_xor(ps, 2, 32);
      ps += __shfl_xor(ps, 4, 32);
      ps += __shfl_xor(ps, 8, 32);
      lrow[r] = lrow[r] * alpha + ps;
      mrow[r] = mnew;
#pragma unroll
      for (int nt = 0; nt < 4; ++nt) o[nt][r] *= alpha;
    }

    // O += P(16x64) x V(64x64)
    const v16bf pa0 = load_a_tile(lP[wave], 64);
    const v16bf pa1 = load_a_tile(lP[wave] + 32, 64);
#pragma unroll
    for (int nt = 0; nt < 4; ++nt) {
      o[nt] = wmma_bf16(pa0, load_b_tile(lVt[buf] + nt * 16 * 64, 64), o[nt]);
      o[nt] = wmma_bf16(pa1, load_b_tile(lVt[buf] + nt * 16 * 64 + 32, 64), o[nt]);
    }

    if (more) scatterV(lVt[buf ^ 1]);
    async_join();
    __syncthreads();
  }

#pragma unroll
  for (int r = 0; r < 8; ++r) {
    const float inv = 1.0f / lrow[r];
    const int s = q0 + wave * 16 + r + half * 8;
#pragma unroll
    for (int nt = 0; nt < 4; ++nt) {
      const int col = h * DK + nt * 16 + nn;
      concat[((size_t)b * SEQ + s) * DM + col] = o[nt][r] * inv;
    }
  }
}

// ============================================================================
// Kernel 3: output projection  out = concat @ Wo^T + bo  (f32 out)
// ============================================================================
__global__ __launch_bounds__(128)
void out_proj_kernel(const float* __restrict__ X, const float* __restrict__ W,
                     const float* __restrict__ bias, float* __restrict__ out) {
  __shared__ __bf16 lA[128 * 32];
  __shared__ __bf16 lW[64 * 32];
  const int tid = threadIdx.x, wave = tid >> 5;
  const int lane = tid & 31, nn = lane & 15, half = lane >> 4;
  const int m0 = blockIdx.x * 128;
  const int n0 = blockIdx.y * 64;

  gemm_body(X, W, m0, n0, lA, lW, [&](v8f acc[2][4]) {
#pragma unroll
    for (int mt = 0; mt < 2; ++mt)
#pragma unroll
      for (int nt = 0; nt < 4; ++nt)
#pragma unroll
        for (int r = 0; r < 8; ++r) {
          const int gm = m0 + wave * 32 + mt * 16 + r + half * 8;
          const int gn = n0 + nt * 16 + nn;
          out[(size_t)gm * DM + gn] = acc[mt][nt][r] + bias[gn];
        }
  });
}

extern "C" void kernel_launch(void* const* d_in, const int* in_sizes, int n_in,
                              void* d_out, int out_size, void* d_ws, size_t ws_size,
                              hipStream_t stream) {
  (void)in_sizes; (void)n_in; (void)out_size; (void)ws_size;
  const float* q  = (const float*)d_in[0];
  const float* k  = (const float*)d_in[1];
  const float* v  = (const float*)d_in[2];
  const float* Wk = (const float*)d_in[3];
  const float* bk = (const float*)d_in[4];
  const float* Wo = (const float*)d_in[5];
  const float* bo = (const float*)d_in[6];
  float* out = (float*)d_out;

  // workspace: qh/kh/vh bf16 [B,H,S,Dk] (6 MB each) + concat f32 (12.6 MB)
  const size_t elems = (size_t)BATCH * SEQ * DM;  // 3,145,728
  __bf16* qh = (__bf16*)d_ws;
  __bf16* kh = qh + elems;
  __bf16* vh = kh + elems;
  float* concat = (float*)(vh + elems);

  dim3 gp(MROWS / 128, DM / 64, 3);
  proj_qkv_kernel<<<gp, 128, 0, stream>>>(q, k, v, Wk, bk, qh, kh, vh);

  dim3 ga(SEQ / 64, BATCH * HEADS);
  attn_kernel<<<ga, 128, 0, stream>>>(qh, kh, vh, concat);

  dim3 go(MROWS / 128, DM / 64);
  out_proj_kernel<<<go, 128, 0, stream>>>(concat, Wo, bo, out);
}
